// Head_24747601560243
// MI455X (gfx1250) — matrix operand order repaired and checked by reading, not compiled, not run on hardware
//
#include <hip/hip_runtime.h>

// Causal single-head attention, flash-attention style, bf16 WMMA on gfx1250.
// inputs: x[8,2048,768] f32, Wk/Wq/Wv [768,64] f32; output [8,2048,64] f32.
// workspace: Qb,Kb bf16 [B*T,64]; Vt bf16 [B,64,T]; Wt bf16 [3,64,768] (~6.3MB)

#define B_SZ   8
#define T_SEQ  2048
#define C_DIM  768
#define H_DIM  64

typedef __attribute__((ext_vector_type(16))) __bf16 bf16x16;
typedef __attribute__((ext_vector_type(2)))  __bf16 bf16x2;
typedef __attribute__((ext_vector_type(8)))  float  f32x8;

union Frag {
  unsigned int u[8];
  bf16x16 v;
};

// f32 -> bf16 with hardware cvt (v_cvt_pk_bf16_f32) when available.
__device__ __forceinline__ unsigned int pack2bf(float lo, float hi) {
#if __has_builtin(__builtin_amdgcn_cvt_pk_bf16_f32)
  bf16x2 p = __builtin_amdgcn_cvt_pk_bf16_f32(lo, hi);
  return __builtin_bit_cast(unsigned int, p);
#else
  bf16x2 p = { (__bf16)lo, (__bf16)hi };
  return __builtin_bit_cast(unsigned int, p);
#endif
}
__device__ __forceinline__ unsigned short f2bf(float f) {
  __bf16 h = (__bf16)f;
  return __builtin_bit_cast(unsigned short, h);
}
__device__ __forceinline__ float fast_exp2(float x) {
#if __has_builtin(__builtin_amdgcn_exp2f)
  return __builtin_amdgcn_exp2f(x);
#else
  return exp2f(x);
#endif
}
__device__ __forceinline__ float fast_rcp(float x) {
#if __has_builtin(__builtin_amdgcn_rcpf)
  return __builtin_amdgcn_rcpf(x);
#else
  return 1.0f / x;
#endif
}
__device__ __forceinline__ f32x8 wmma_bf16(const Frag& a, const Frag& b, f32x8 c) {
  return __builtin_amdgcn_wmma_f32_16x16x32_bf16(false, a.v, false, b.v,
                                                 (short)0, c, false, false);
}

// ---------------- kernel 1: weights -> bf16, transposed [3][64][768] --------
__global__ __launch_bounds__(256) void wconv_kernel(const float* __restrict__ Wk,
                                                    const float* __restrict__ Wq,
                                                    const float* __restrict__ Wv,
                                                    unsigned short* __restrict__ Wt) {
  int idx = blockIdx.x * 256 + threadIdx.x;
  if (idx >= 3 * C_DIM * H_DIM) return;
  int w = idx / (C_DIM * H_DIM);
  int r = idx % (C_DIM * H_DIM);
  int k = r / H_DIM;
  int n = r % H_DIM;
  const float* W = (w == 0) ? Wq : (w == 1) ? Wk : Wv;
  Wt[((size_t)w * H_DIM + n) * C_DIM + k] = f2bf(W[k * H_DIM + n]);
}

// ---------------- kernel 2: Q,K,V projections (x @ W), bf16 WMMA ------------
// grid.x = B*T/128; 256 threads = 8 waves; wave owns 16 rows x 64 cols x 3 mats
__global__ __launch_bounds__(256) void proj_kernel(const float* __restrict__ x,
                                                   const unsigned short* __restrict__ Wt,
                                                   unsigned short* __restrict__ Qb,
                                                   unsigned short* __restrict__ Kb,
                                                   unsigned short* __restrict__ Vt) {
  const int wave  = threadIdx.x >> 5;
  const int lane  = threadIdx.x & 31;
  const int nlo   = lane & 15;
  const int lhalf = lane >> 4;
  const long row0 = (long)blockIdx.x * 128 + wave * 16;

  f32x8 acc[3][4];
  #pragma unroll
  for (int w = 0; w < 3; ++w)
    #pragma unroll
    for (int c = 0; c < 4; ++c)
      acc[w][c] = (f32x8){0.f,0.f,0.f,0.f,0.f,0.f,0.f,0.f};

  const float* xrow = x + (row0 + nlo) * C_DIM;   // A-fragment row = lane&15
  for (int kc = 0; kc < C_DIM; kc += 32) {
    Frag a;
    #pragma unroll
    for (int vv = 0; vv < 8; ++vv) {
      int k0 = kc + ((vv >> 2) << 4) + (lhalf << 3) + ((vv & 3) << 1);
      float2 f = *(const float2*)(xrow + k0);
      a.u[vv] = pack2bf(f.x, f.y);
    }
    #pragma unroll
    for (int w = 0; w < 3; ++w) {
      const unsigned short* Wm = Wt + (size_t)w * H_DIM * C_DIM;
      #pragma unroll
      for (int c = 0; c < 4; ++c) {
        Frag bm;
        #pragma unroll
        for (int vv = 0; vv < 8; ++vv) {
          int k0 = kc + ((vv >> 2) << 4) + (lhalf << 3) + ((vv & 3) << 1);
          bm.u[vv] = *(const unsigned int*)(Wm + (size_t)(c * 16 + nlo) * C_DIM + k0);
        }
        acc[w][c] = wmma_bf16(a, bm, acc[w][c]);
      }
    }
  }
  // C-layout store: VGPR vv holds row (lhalf*8+vv), col = lane&15 (+16*c)
  #pragma unroll
  for (int vv = 0; vv < 8; ++vv) {
    long r = row0 + (lhalf << 3) + vv;
    int  b = (int)(r / T_SEQ);
    int  t = (int)(r % T_SEQ);
    #pragma unroll
    for (int c = 0; c < 4; ++c) {
      int n = c * 16 + nlo;
      Qb[r * H_DIM + n] = f2bf(acc[0][c][vv]);
      Kb[r * H_DIM + n] = f2bf(acc[1][c][vv]);
      Vt[((long)b * H_DIM + n) * T_SEQ + t] = f2bf(acc[2][c][vv]);   // transposed V
    }
  }
}

// ---------------- kernel 3: flash attention, causal, online softmax ---------
// grid = (T/128, B); 8 waves/block; wave owns 16 query rows
__global__ __launch_bounds__(256) void attn_kernel(const unsigned short* __restrict__ Qb,
                                                   const unsigned short* __restrict__ Kb,
                                                   const unsigned short* __restrict__ Vt,
                                                   float* __restrict__ out) {
  __shared__ unsigned short plds[8][16 * 32];     // per-wave 16x32 P tile (bf16)
  const int wave  = threadIdx.x >> 5;
  const int lane  = threadIdx.x & 31;
  const int nlo   = lane & 15;
  const int lhalf = lane >> 4;
  const int b     = blockIdx.y;
  const int q0    = blockIdx.x * 128 + wave * 16;
  const long qrow = (long)b * T_SEQ + q0;
  // 0.125 (head scale) * log2(e): softmax runs in exp2 domain
  const float ESC = 0.18033688011112042f;

  // Q A-fragments (16 rows x 64 K, split into two 16x32 frags)
  Frag qa[2];
  #pragma unroll
  for (int h = 0; h < 2; ++h)
    #pragma unroll
    for (int vv = 0; vv < 8; ++vv) {
      int k0 = h * 32 + ((vv >> 2) << 4) + (lhalf << 3) + ((vv & 3) << 1);
      qa[h].u[vv] = *(const unsigned int*)(Qb + (qrow + nlo) * H_DIM + k0);
    }

  f32x8 o[4];
  #pragma unroll
  for (int c = 0; c < 4; ++c) o[c] = (f32x8){0.f,0.f,0.f,0.f,0.f,0.f,0.f,0.f};
  float mrun[8], lrun[8];
  #pragma unroll
  for (int vv = 0; vv < 8; ++vv) { mrun[vv] = -__builtin_inff(); lrun[vv] = 0.f; }

  const unsigned short* Kbase = Kb + (long)b * T_SEQ * H_DIM;
  const unsigned short* Vbase = Vt + (long)b * H_DIM * T_SEQ;

  for (int j = 0; j < q0 + 16; j += 32) {         // 32 keys per chunk
    // --- S = Q @ K^T : two 16x16 score tiles ---
    f32x8 s0 = (f32x8){0.f,0.f,0.f,0.f,0.f,0.f,0.f,0.f};
    f32x8 s1 = s0;
    #pragma unroll
    for (int h = 0; h < 2; ++h) {                 // K-dim 64 = 2 x 32
      Frag kb0, kb1;
      #pragma unroll
      for (int vv = 0; vv < 8; ++vv) {
        int k0 = h * 32 + ((vv >> 2) << 4) + (lhalf << 3) + ((vv & 3) << 1);
        kb0.u[vv] = *(const unsigned int*)(Kbase + (long)(j + nlo) * H_DIM + k0);
        kb1.u[vv] = *(const unsigned int*)(Kbase + (long)(j + 16 + nlo) * H_DIM + k0);
      }
      s0 = wmma_bf16(qa[h], kb0, s0);
      s1 = wmma_bf16(qa[h], kb1, s1);
    }
    if (j + 32 < q0 + 16)
      __builtin_prefetch(Kbase + (long)(j + 32 + nlo) * H_DIM, 0, 1);

    // --- V B-fragments issued early: latency overlaps softmax, and they stay
    //     above the DS-fence memory clobber below ---
    Frag vb[4];
    #pragma unroll
    for (int c = 0; c < 4; ++c)
      #pragma unroll
      for (int vv = 0; vv < 8; ++vv) {
        int k0 = ((vv >> 2) << 4) + (lhalf << 3) + ((vv & 3) << 1);
        vb[c].u[vv] = *(const unsigned int*)(Vbase + (long)(c * 16 + nlo) * T_SEQ + j + k0);
      }

    // --- scale, causal mask, online softmax (fp32, exp2 domain) ---
    #pragma unroll
    for (int vv = 0; vv < 8; ++vv) {
      const int mr = (lhalf << 3) + vv;           // row within strip
      const int qg = q0 + mr;                     // global query index
      float e0 = s0[vv] * ESC;
      float e1 = s1[vv] * ESC;
      if (j + nlo > qg)      e0 = -__builtin_inff();
      if (j + 16 + nlo > qg) e1 = -__builtin_inff();
      float cm = fmaxf(e0, e1);
      #pragma unroll
      for (int off = 1; off <= 8; off <<= 1)      // reduce over 16-lane row group
        cm = fmaxf(cm, __shfl_xor(cm, off, 32));
      float mnew  = fmaxf(mrun[vv], cm);
      float alpha = fast_exp2(mrun[vv] - mnew);
      float p0 = fast_exp2(e0 - mnew);
      float p1 = fast_exp2(e1 - mnew);
      float rs = p0 + p1;
      #pragma unroll
      for (int off = 1; off <= 8; off <<= 1)
        rs += __shfl_xor(rs, off, 32);
      lrun[vv] = lrun[vv] * alpha + rs;
      mrun[vv] = mnew;
      o[0][vv] *= alpha; o[1][vv] *= alpha; o[2][vv] *= alpha; o[3][vv] *= alpha;
      plds[wave][mr * 32 + nlo]      = f2bf(p0); // C-layout -> LDS row-major
      plds[wave][mr * 32 + 16 + nlo] = f2bf(p1);
    }
    asm volatile("s_wait_dscnt 0" ::: "memory");  // wave-local LDS RAW fence

    // --- P A-fragment from LDS (row = lane&15) ---
    Frag pa;
    #pragma unroll
    for (int vv = 0; vv < 8; ++vv) {
      int k0 = ((vv >> 2) << 4) + (lhalf << 3) + ((vv & 3) << 1);
      pa.u[vv] = *(const unsigned int*)(&plds[wave][nlo * 32 + k0]);
    }
    // --- O += P @ V ---
    #pragma unroll
    for (int c = 0; c < 4; ++c)
      o[c] = wmma_bf16(pa, vb[c], o[c]);
    asm volatile("s_wait_dscnt 0" ::: "memory");  // WAR fence before next chunk's stores
  }

  #pragma unroll
  for (int vv = 0; vv < 8; ++vv) {
    int mr = (lhalf << 3) + vv;
    float inv = fast_rcp(lrun[vv]);
    #pragma unroll
    for (int c = 0; c < 4; ++c)
      out[(qrow + mr) * H_DIM + c * 16 + nlo] = o[c][vv] * inv;
  }
}

// ---------------------------------------------------------------------------
extern "C" void kernel_launch(void* const* d_in, const int* in_sizes, int n_in,
                              void* d_out, int out_size, void* d_ws, size_t ws_size,
                              hipStream_t stream) {
  const float* x  = (const float*)d_in[0];
  const float* Wk = (const float*)d_in[1];
  const float* Wq = (const float*)d_in[2];
  const float* Wv = (const float*)d_in[3];
  float* out = (float*)d_out;

  unsigned short* Qb = (unsigned short*)d_ws;                      // [B*T,64] bf16
  unsigned short* Kb = Qb + (size_t)B_SZ * T_SEQ * H_DIM;          // [B*T,64] bf16
  unsigned short* Vt = Kb + (size_t)B_SZ * T_SEQ * H_DIM;          // [B,64,T] bf16
  unsigned short* Wt = Vt + (size_t)B_SZ * T_SEQ * H_DIM;          // [3,64,768] bf16

  wconv_kernel<<<dim3((3 * C_DIM * H_DIM + 255) / 256), dim3(256), 0, stream>>>(Wk, Wq, Wv, Wt);
  proj_kernel<<<dim3(B_SZ * T_SEQ / 128), dim3(256), 0, stream>>>(x, Wt, Qb, Kb, Vt);
  attn_kernel<<<dim3(T_SEQ / 128, B_SZ), dim3(256), 0, stream>>>(Qb, Kb, Vt, out);
}
